// CaputoEncoder_89086211654059
// MI455X (gfx1250) — matrix operand order repaired and checked by reading, compiled, not run on hardware
//
#include <hip/hip_runtime.h>
#include <math.h>

// ---------------------------------------------------------------------------
// CDNA5 (gfx1250): Caputo features + 2-layer LSTM + projection.
// GEMMs: V_WMMA_F32_16X16X32_F16, 4x4 register-tiled (64x64 per wave).
// LSTM step: async global->LDS staging of the shared A panel.
// ---------------------------------------------------------------------------

typedef __attribute__((ext_vector_type(16))) _Float16 v16h;
typedef __attribute__((ext_vector_type(8)))  float    v8f;
typedef __attribute__((__vector_size__(16))) int      i32x4;

#define AS_GLOBAL __attribute__((address_space(1)))
#define AS_LDS    __attribute__((address_space(3)))

static constexpr int Bsz = 64;    // batch
static constexpr int T   = 512;   // time steps
static constexpr int H   = 512;   // LSTM hidden
static constexpr int G   = 2048;  // 4*H gates
static constexpr int Fp  = 512;   // padded feature width (500 -> 512)

#if __has_builtin(__builtin_amdgcn_global_load_async_to_lds_b128) && \
    __has_builtin(__builtin_amdgcn_s_wait_asynccnt)
#define HAVE_ASYNC_LDS 1
#else
#define HAVE_ASYNC_LDS 0
#endif

__device__ __forceinline__ v8f wmma_f16(v16h a, v16h b, v8f c) {
  // 8 args: (neg_a, A, neg_b, B, c_mod, C, reuse_a, reuse_b)
  return __builtin_amdgcn_wmma_f32_16x16x32_f16(false, a, false, b, (short)0, c,
                                                false, false);
}

// A-matrix tile (16x32 f16, MxK) from row-major [.., lda] f16 matrix.
// Contiguous k order (matches load_bT16): lane m=lane&15, half=lane>>4; elem e -> K = half*16 + e
__device__ __forceinline__ v16h load_a16(const _Float16* __restrict__ A, int lda,
                                         int m0, int k0, int lane) {
  int m    = m0 + (lane & 15);
  int half = lane >> 4;
  const _Float16* p = A + (size_t)m * lda + k0 + half * 16;
  v16h r;
#pragma unroll
  for (int e = 0; e < 16; ++e) r[e] = p[e];
  return r;
}

// B-matrix tile (32x16 f16, KxN) where B[k][n] = W[n][k], W row-major [N, ldk].
// Contiguous k order (matches load_a16): lane n=lane&15, khalf=lane>>4; elem e -> K = khalf*16 + e
__device__ __forceinline__ v16h load_bT16(const _Float16* __restrict__ W, int ldk,
                                          int n0, int k0, int lane) {
  int n     = n0 + (lane & 15);
  int khalf = lane >> 4;
  const _Float16* p = W + (size_t)n * ldk + k0 + khalf * 16;
  v16h r;
#pragma unroll
  for (int e = 0; e < 16; ++e) r[e] = p[e];
  return r;
}

__device__ __forceinline__ float sigm(float x) { return 1.0f / (1.0f + expf(-x)); }

// ---------------------------------------------------------------------------
// Prep kernels
// ---------------------------------------------------------------------------

__global__ void k_zero_h(_Float16* p, long long n) {
  long long i = (long long)blockIdx.x * blockDim.x + threadIdx.x;
  if (i < n) p[i] = (_Float16)0.0f;
}
__global__ void k_zero_f(float* p, long long n) {
  long long i = (long long)blockIdx.x * blockDim.x + threadIdx.x;
  if (i < n) p[i] = 0.0f;
}

// W[a][t][j] = (t-j+1e-6)^(-alpha) for t>j else 0 ; rowsum[a][t] = sum_j W
__global__ void k_prep_caputo(_Float16* __restrict__ Wcap, float* __restrict__ rowsum) {
  int a = blockIdx.x >> 9;
  int t = blockIdx.x & 511;
  float alpha = a ? 1.0f : 0.5f;
  __shared__ float sbuf[256];
  float s = 0.0f;
  for (int j = threadIdx.x; j < T; j += 256) {
    int diff = t - j;
    float w = 0.0f;
    if (diff > 0) w = powf((float)diff + 1e-6f, -alpha);
    Wcap[((size_t)a * T + t) * T + j] = (_Float16)w;
    s += w;
  }
  sbuf[threadIdx.x] = s;
  __syncthreads();
  for (int st = 128; st > 0; st >>= 1) {
    if ((int)threadIdx.x < st) sbuf[threadIdx.x] += sbuf[threadIdx.x + st];
    __syncthreads();
  }
  if (threadIdx.x == 0) rowsum[a * T + t] = sbuf[0];
}

// xT[b][n][j] = f16(x[b][j][n]) for n<250 else 0.  (256 x 512 per batch)
__global__ void k_transpose_x(const float* __restrict__ x, _Float16* __restrict__ xT) {
  long long i = (long long)blockIdx.x * blockDim.x + threadIdx.x;
  if (i >= (long long)Bsz * 256 * T) return;
  int j = (int)(i & 511);
  int n = (int)((i >> 9) & 255);
  int b = (int)(i >> 17);
  _Float16 v = (_Float16)0.0f;
  if (n < 250) v = (_Float16)x[((size_t)b * T + j) * 250 + n];
  xT[i] = v;
}

// dst[r][c] (f16, R x Cp) = src[r][c] (f32, R x C), zero padded to Cp.
__global__ void k_pack16(const float* __restrict__ src, _Float16* __restrict__ dst,
                         int R, int C, int Cp) {
  long long i = (long long)blockIdx.x * blockDim.x + threadIdx.x;
  if (i >= (long long)R * Cp) return;
  int c = (int)(i % Cp);
  int r = (int)(i / Cp);
  dst[i] = (c < C) ? (_Float16)src[(size_t)r * C + c] : (_Float16)0.0f;
}

__global__ void k_bias2(const float* a, const float* b, float* o, int n) {
  int i = blockIdx.x * blockDim.x + threadIdx.x;
  if (i < n) o[i] = a[i] + b[i];
}

// ---------------------------------------------------------------------------
// 4x4 register-tiled WMMA core: one wave -> 64x64 output patch.
// B tiles are loaded one at a time to keep the live set at acc+A+1xB.
// ---------------------------------------------------------------------------
struct Acc44 { v8f a[4][4]; };

__device__ __forceinline__ void gemm64x64(Acc44& acc,
                                          const _Float16* __restrict__ A, int lda,
                                          const _Float16* __restrict__ W, int ldk,
                                          int m0, int n0, int K, int lane) {
  for (int k0 = 0; k0 < K; k0 += 32) {
    if (k0 + 32 < K) {  // -> global_prefetch_b8
      __builtin_prefetch(A + (size_t)(m0 + (lane & 15)) * lda + k0 + 32);
      __builtin_prefetch(W + (size_t)(n0 + (lane & 15)) * ldk + k0 + 32);
    }
    v16h av[4];
#pragma unroll
    for (int i = 0; i < 4; ++i) av[i] = load_a16(A, lda, m0 + 16 * i, k0, lane);
#pragma unroll
    for (int j = 0; j < 4; ++j) {
      v16h bv = load_bT16(W, ldk, n0 + 16 * j, k0, lane);
#pragma unroll
      for (int i = 0; i < 4; ++i)
        acc.a[i][j] = wmma_f16(av[i], bv, acc.a[i][j]);
    }
  }
}

// ---------------------------------------------------------------------------
// Caputo feature GEMM: C[t,n] = sum_j Wcap[a][t][j] * x[b][j][n]  (WMMA)
// feats[b*T+t][a*250+n] = coef_a * (x[b][t][n]*rowsum[a][t] - C[t,n])
// Wave -> 64(t) x 64(n) patch; waves per (b,a) = 8 x 4.
// ---------------------------------------------------------------------------
__global__ void __launch_bounds__(128, 1)
k_caputo(const _Float16* __restrict__ Wcap, const float* __restrict__ rowsum,
         const _Float16* __restrict__ xT, const float* __restrict__ x,
         _Float16* __restrict__ feats, float coef0, float coef1) {
  int wid  = blockIdx.x * (blockDim.x >> 5) + (threadIdx.x >> 5);
  int lane = threadIdx.x & 31;
  if (wid >= 2 * Bsz * 8 * 4) return;
  int tn = wid & 3;
  int tm = (wid >> 2) & 7;
  int ba = wid >> 5;
  int a  = ba & 1;
  int b  = ba >> 1;
  float coef = a ? coef1 : coef0;
  const _Float16* A  = Wcap + (size_t)a * T * T;  // 512 x 512
  const _Float16* Bm = xT + (size_t)b * 256 * T;  // 256 x 512 (n-major)
  int m0 = tm * 64, n0 = tn * 64;
  Acc44 acc = {};
  gemm64x64(acc, A, T, Bm, T, m0, n0, T, lane);
  int half = lane >> 4;
#pragma unroll
  for (int j = 0; j < 4; ++j) {
    int n = n0 + j * 16 + (lane & 15);
    if (n >= 250) continue;  // epilogue-only divergence
#pragma unroll
    for (int i = 0; i < 4; ++i)
#pragma unroll
      for (int r = 0; r < 8; ++r) {
        int t = m0 + i * 16 + r + half * 8;
        float xv = x[((size_t)b * T + t) * 250 + n];
        float v  = coef * (xv * rowsum[a * T + t] - acc.a[i][j][r]);
        feats[((size_t)b * T + t) * Fp + a * 250 + n] = (_Float16)v;
      }
  }
}

// ---------------------------------------------------------------------------
// Generic GEMM: out[M,N] = A[M,K](f16) @ W[N,K]^T(f16) + bias, optional relu.
// M, N multiples of 64.
// ---------------------------------------------------------------------------
__global__ void __launch_bounds__(128, 1)
k_gemm16(const _Float16* __restrict__ A, int lda,
         const _Float16* __restrict__ W, int ldk,
         const float* __restrict__ bias,
         float* __restrict__ outF, _Float16* __restrict__ outH,
         int M, int N, int K, int relu) {
  int wid  = blockIdx.x * (blockDim.x >> 5) + (threadIdx.x >> 5);
  int lane = threadIdx.x & 31;
  int ntn = N >> 6, ntm = M >> 6;
  if (wid >= ntm * ntn) return;
  int tm = wid / ntn, tn = wid % ntn;
  int m0 = tm * 64, n0 = tn * 64;
  Acc44 acc = {};
  gemm64x64(acc, A, lda, W, ldk, m0, n0, K, lane);
  int half = lane >> 4;
#pragma unroll
  for (int j = 0; j < 4; ++j) {
    int n = n0 + j * 16 + (lane & 15);
    float bvl = bias ? bias[n] : 0.0f;
#pragma unroll
    for (int i = 0; i < 4; ++i)
#pragma unroll
      for (int r = 0; r < 8; ++r) {
        int m = m0 + i * 16 + r + half * 8;
        float v = acc.a[i][j][r] + bvl;
        if (relu) v = fmaxf(v, 0.0f);
        if (outF) outF[(size_t)m * N + n] = v;
        if (outH) outH[(size_t)m * N + n] = (_Float16)v;
      }
  }
}

// ---------------------------------------------------------------------------
// One LSTM timestep, fully fused. A block (4 waves) shares one 16(batch)-row
// A panel of h_prev: staged global->LDS with ASYNC loads, then each wave runs
// 4 gate WMMA accumulators against Whh and does the cell update.
// ---------------------------------------------------------------------------
__global__ void __launch_bounds__(128, 1)
k_lstm_step(const float* __restrict__ xW,      // [B*T, 2048], row b*T+t
            const _Float16* __restrict__ Whh,  // [2048, 512] f16
            float* __restrict__ c,             // [64, 512] f32
            const _Float16* __restrict__ hin,  // [64, 512] f16
            _Float16* __restrict__ hout,       // [64, 512] f16
            _Float16* __restrict__ hseq,       // [B*T, 512] or null
            int t) {
  __shared__ _Float16 sA[16 * H];  // 16 KB: one 16-row A panel
  int lane = threadIdx.x & 31;
  int gwid = blockIdx.x * (blockDim.x >> 5) + (threadIdx.x >> 5);
  if (gwid >= (Bsz / 16) * (H / 16)) return;
  int tm = gwid >> 5, tn = gwid & 31;  // all 4 waves of a block share tm
  int m0 = tm * 16, n0 = tn * 16;

  // Stage hin rows [m0, m0+16) x 512 halves (16 KB) into LDS.
  const _Float16* gsrc = hin + (size_t)m0 * H;
#if HAVE_ASYNC_LDS
  for (int i = threadIdx.x; i < (16 * H) / 8; i += 128) {
    __builtin_amdgcn_global_load_async_to_lds_b128(
        (AS_GLOBAL i32x4*)(gsrc + i * 8),
        (AS_LDS i32x4*)(sA + i * 8), 0, 0);
  }
  __builtin_amdgcn_s_wait_asynccnt(0);
#else
  for (int i = threadIdx.x; i < (16 * H) / 8; i += 128) {
    *(uint4*)(sA + i * 8) = *(const uint4*)(gsrc + i * 8);
  }
#endif
  __syncthreads();

  v8f ai = {}, af = {}, ag = {}, ao = {};
  for (int k0 = 0; k0 < H; k0 += 32) {
    v16h av = load_a16(sA, H, 0, k0, lane);  // ds_load from staged panel
    ai = wmma_f16(av, load_bT16(Whh, H, 0 * H + n0, k0, lane), ai);
    af = wmma_f16(av, load_bT16(Whh, H, 1 * H + n0, k0, lane), af);
    ag = wmma_f16(av, load_bT16(Whh, H, 2 * H + n0, k0, lane), ag);
    ao = wmma_f16(av, load_bT16(Whh, H, 3 * H + n0, k0, lane), ao);
  }
  int n = n0 + (lane & 15);
  int half = lane >> 4;
#pragma unroll
  for (int r = 0; r < 8; ++r) {
    int m = m0 + r + half * 8;
    const float* xr = xW + ((size_t)m * T + t) * G;
    float gi = sigm(ai[r] + xr[0 * H + n]);
    float gf = sigm(af[r] + xr[1 * H + n]);
    float gg = tanhf(ag[r] + xr[2 * H + n]);
    float go = sigm(ao[r] + xr[3 * H + n]);
    float cc = gf * c[(size_t)m * H + n] + gi * gg;
    c[(size_t)m * H + n] = cc;
    float hh = go * tanhf(cc);
    hout[(size_t)m * H + n] = (_Float16)hh;
    if (hseq) hseq[((size_t)m * T + t) * H + n] = (_Float16)hh;
  }
}

// ---------------------------------------------------------------------------
// Host orchestration
// ---------------------------------------------------------------------------
extern "C" void kernel_launch(void* const* d_in, const int* in_sizes, int n_in,
                              void* d_out, int out_size, void* d_ws, size_t ws_size,
                              hipStream_t stream) {
  (void)in_sizes; (void)n_in; (void)out_size; (void)ws_size;
  const float* x    = (const float*)d_in[0];
  const float* Wih0 = (const float*)d_in[1];
  const float* Whh0 = (const float*)d_in[2];
  const float* bih0 = (const float*)d_in[3];
  const float* bhh0 = (const float*)d_in[4];
  const float* Wih1 = (const float*)d_in[5];
  const float* Whh1 = (const float*)d_in[6];
  const float* bih1 = (const float*)d_in[7];
  const float* bhh1 = (const float*)d_in[8];
  const float* Wout = (const float*)d_in[9];
  const float* bout = (const float*)d_in[10];

  char* ws = (char*)d_ws;
  size_t off = 0;
  auto take = [&](size_t bytes) -> char* {
    char* p = ws + off;
    off = (off + bytes + 255) & ~(size_t)255;
    return p;
  };
  _Float16* Wcap   = (_Float16*)take((size_t)2 * T * T * 2);
  float*    rowsum = (float*)   take((size_t)2 * T * 4);
  _Float16* xTh    = (_Float16*)take((size_t)Bsz * 256 * T * 2);
  _Float16* feats  = (_Float16*)take((size_t)Bsz * T * Fp * 2);
  _Float16* Wih0p  = (_Float16*)take((size_t)G * Fp * 2);
  _Float16* Whh0h  = (_Float16*)take((size_t)G * H * 2);
  _Float16* Wih1h  = (_Float16*)take((size_t)G * H * 2);
  _Float16* Whh1h  = (_Float16*)take((size_t)G * H * 2);
  _Float16* Wouth  = (_Float16*)take((size_t)1024 * H * 2);
  float*    bias0  = (float*)   take((size_t)G * 4);
  float*    bias1  = (float*)   take((size_t)G * 4);
  float*    xW     = (float*)   take((size_t)Bsz * T * G * 4);   // 268 MB
  _Float16* hseq   = (_Float16*)take((size_t)Bsz * T * H * 2);
  float*    cst    = (float*)   take((size_t)Bsz * H * 4);
  _Float16* hbuf0  = (_Float16*)take((size_t)Bsz * H * 2);
  _Float16* hbuf1  = (_Float16*)take((size_t)Bsz * H * 2);

  const float coef0 = 0.5641895835477563f;  // 1/Gamma(1-0.5)
  const float coef1 = 0.0f;                 // 1/Gamma(0) -> 0 by reference's rule

  // --- prep ---
  k_prep_caputo<<<1024, 256, 0, stream>>>(Wcap, rowsum);
  {
    long long n = (long long)Bsz * 256 * T;
    k_transpose_x<<<(unsigned)((n + 255) / 256), 256, 0, stream>>>(x, xTh);
  }
  k_pack16<<<(2048 * 512) / 256, 256, 0, stream>>>(Wih0, Wih0p, G, 500, Fp);
  k_pack16<<<(2048 * 512) / 256, 256, 0, stream>>>(Whh0, Whh0h, G, H, H);
  k_pack16<<<(2048 * 512) / 256, 256, 0, stream>>>(Wih1, Wih1h, G, H, H);
  k_pack16<<<(2048 * 512) / 256, 256, 0, stream>>>(Whh1, Whh1h, G, H, H);
  k_pack16<<<(1024 * 512) / 256, 256, 0, stream>>>(Wout, Wouth, 1024, H, H);
  k_bias2<<<8, 256, 0, stream>>>(bih0, bhh0, bias0, G);
  k_bias2<<<8, 256, 0, stream>>>(bih1, bhh1, bias1, G);
  {
    long long n = (long long)Bsz * T * Fp;  // zero feats (covers pad cols 500..511)
    k_zero_h<<<(unsigned)((n + 255) / 256), 256, 0, stream>>>(feats, n);
  }

  // --- Caputo features (WMMA, 64x64 per wave) ---
  {
    int waves = 2 * Bsz * 8 * 4;  // 4096
    k_caputo<<<waves / 4, 128, 0, stream>>>(Wcap, rowsum, xTh, x, feats, coef0, coef1);
  }

  // --- layer 0 input projection: xW = feats @ Wih0^T + (bih0+bhh0) ---
  {
    int waves = (Bsz * T / 64) * (G / 64);  // 16384
    k_gemm16<<<waves / 4, 128, 0, stream>>>(feats, Fp, Wih0p, Fp, bias0,
                                            xW, nullptr, Bsz * T, G, Fp, 0);
  }

  // --- layer 0 recurrent scan ---
  k_zero_f<<<(Bsz * H) / 256, 256, 0, stream>>>(cst, (long long)Bsz * H);
  k_zero_h<<<(Bsz * H) / 256, 256, 0, stream>>>(hbuf0, (long long)Bsz * H);
  _Float16* hb[2] = {hbuf0, hbuf1};
  for (int t = 0; t < T; ++t) {
    k_lstm_step<<<32, 128, 0, stream>>>(xW, Whh0h, cst, hb[t & 1], hb[(t & 1) ^ 1],
                                        hseq, t);
  }

  // --- layer 1 input projection: xW = hseq @ Wih1^T + (bih1+bhh1) ---
  {
    int waves = (Bsz * T / 64) * (G / 64);
    k_gemm16<<<waves / 4, 128, 0, stream>>>(hseq, H, Wih1h, H, bias1,
                                            xW, nullptr, Bsz * T, G, H, 0);
  }

  // --- layer 1 recurrent scan (no hseq output; final h ends in hbuf0) ---
  k_zero_f<<<(Bsz * H) / 256, 256, 0, stream>>>(cst, (long long)Bsz * H);
  k_zero_h<<<(Bsz * H) / 256, 256, 0, stream>>>(hbuf0, (long long)Bsz * H);
  for (int t = 0; t < T; ++t) {
    k_lstm_step<<<32, 128, 0, stream>>>(xW, Whh1h, cst, hb[t & 1], hb[(t & 1) ^ 1],
                                        nullptr, t);
  }
  // t=511 (odd) wrote hb[0] -> final hidden state is hbuf0

  // --- output: relu(h_last @ Wout^T + bout) ---
  {
    int waves = (Bsz / 64) * (1024 / 64);  // 16
    k_gemm16<<<waves / 4, 128, 0, stream>>>(hbuf0, H, Wouth, H, bout,
                                            (float*)d_out, nullptr, Bsz, 1024, H, 1);
  }
}
